// DynamicSparseAttention_13932873908464
// MI455X (gfx1250) — matrix operand order repaired and checked by reading, not compiled, not run on hardware
//
#include <hip/hip_runtime.h>
#include <hip/hip_bf16.h>
#include <cstdint>

typedef __bf16 bf16;
typedef __attribute__((ext_vector_type(16))) __bf16 v16bf;
typedef __attribute__((ext_vector_type(8)))  __bf16 v8bf;
typedef __attribute__((ext_vector_type(8)))  float  v8f;
typedef __attribute__((ext_vector_type(4)))  int    v4i;

#define BATCH   2
#define SEQ     2048
#define HIDDEN  2048
#define NH      16
#define KVH     8
#define HD      128
#define TOPK    819            /* int(2048 * 0.4) — q_len > 512 path */

// ---- gfx1250 async global->LDS path (guarded; falls back to sync copies)
#if defined(__has_builtin)
#  if __has_builtin(__builtin_amdgcn_global_load_async_to_lds_b128) && \
      __has_builtin(__builtin_amdgcn_s_wait_asynccnt)
#    define HAVE_ASYNC_LDS 1
#  endif
#endif
#ifndef HAVE_ASYNC_LDS
#  define HAVE_ASYNC_LDS 0
#endif

static __device__ __forceinline__ v8f wmma_bf16(v16bf a, v16bf b, v8f c) {
  // D = A(16x32 bf16) * B(32x16 bf16) + C(16x16 f32)
  return __builtin_amdgcn_wmma_f32_16x16x32_bf16(false, a, false, b, (short)0, c, false, false);
}

static __device__ __forceinline__ v16bf cat8(v8bf lo, v8bf hi) {
  return __builtin_shufflevector(lo, hi, 0,1,2,3,4,5,6,7,8,9,10,11,12,13,14,15);
}

// ---------------------------------------------------------------- convert
__global__ void f32_to_bf16_kernel(const float* __restrict__ src,
                                   bf16* __restrict__ dst, size_t n) {
  size_t i = (size_t)blockIdx.x * blockDim.x + threadIdx.x;
  size_t stride = (size_t)gridDim.x * blockDim.x;
  for (; i < n; i += stride) dst[i] = (bf16)src[i];
}

// ---------------------------------------------------------------- GEMM
// C[M,N] = A[M,K](bf16,rm) @ B[K,N](bf16,rm) + bias ; tile 128x64, 8 waves.
template <bool OUT_F32>
__global__ __launch_bounds__(256)
void gemm_bf16_kernel(const bf16* __restrict__ A, const bf16* __restrict__ B,
                      const float* __restrict__ bias, void* __restrict__ Cout,
                      int M, int N, int K) {
  __shared__ __align__(32) bf16 ldsA[128 * 32];   // [m][k]
  __shared__ __align__(32) bf16 ldsBt[64 * 32];   // [n][k] (transposed)

  const int m0   = blockIdx.y * 128;
  const int n0   = blockIdx.x * 64;
  const int tid  = threadIdx.x;
  const int wave = tid >> 5;       // 0..7 -> owns rows [wave*16, wave*16+16)
  const int lane = tid & 31;
  const int hf   = lane >> 4;      // half-wave
  const int l16  = lane & 15;

  v8f acc[4] = {v8f{}, v8f{}, v8f{}, v8f{}};

  for (int k0 = 0; k0 < K; k0 += 32) {
    { // stage A 128x32 (8 KB): 32B (16 bf16) per thread
      int row = tid >> 1, seg = tid & 1;
      const bf16* src = A + (size_t)(m0 + row) * K + k0 + seg * 16;
      bf16*       dst = &ldsA[row * 32 + seg * 16];
#if HAVE_ASYNC_LDS
      // ASYNCcnt-tracked DMA copy: frees VGPRs, overlaps with WMMA issue
      __builtin_amdgcn_global_load_async_to_lds_b128((v4i*)src, (v4i*)dst, 0, 0);
      __builtin_amdgcn_global_load_async_to_lds_b128((v4i*)(src + 8), (v4i*)(dst + 8), 0, 0);
#else
      ((uint4*)dst)[0] = ((const uint4*)src)[0];
      ((uint4*)dst)[1] = ((const uint4*)src)[1];
#endif
      if (k0 + 32 < K) __builtin_prefetch(src + 32, 0, 0);
    }
    { // stage B 32x64 transposed into ldsBt[n][k] (needs reorder -> manual)
      int kk = tid >> 3, seg = tid & 7;
      const bf16* src = B + (size_t)(k0 + kk) * N + n0 + seg * 8;
      uint4 raw = *(const uint4*)src;
      const bf16* tp = reinterpret_cast<const bf16*>(&raw);
#pragma unroll
      for (int j = 0; j < 8; ++j) ldsBt[(seg * 8 + j) * 32 + kk] = tp[j];
      if (k0 + 32 < K) __builtin_prefetch(src + (size_t)32 * N, 0, 0);
    }
#if HAVE_ASYNC_LDS
    __builtin_amdgcn_s_wait_asynccnt(0);
#endif
    __syncthreads();

    // A fragment (16x32): lane=m, VGPR0..3 <- k=hf*8+0..7, VGPR4..7 <- k=16+hf*8+0..7
    v8bf alo = *(const v8bf*)&ldsA[(wave * 16 + l16) * 32 + hf * 8];
    v8bf ahi = *(const v8bf*)&ldsA[(wave * 16 + l16) * 32 + 16 + hf * 8];
    v16bf afrag = cat8(alo, ahi);
#pragma unroll
    for (int t = 0; t < 4; ++t) {
      // B fragment (32x16): lane col n=t*16+l16, k = hf*8+0..7, 16+hf*8+0..7
      v16bf bfrag = cat8(*(const v8bf*)&ldsBt[(t * 16 + l16) * 32 + hf * 8],
                         *(const v8bf*)&ldsBt[(t * 16 + l16) * 32 + 16 + hf * 8]);
      acc[t] = wmma_bf16(afrag, bfrag, acc[t]);
    }
    __syncthreads();
  }

#pragma unroll
  for (int t = 0; t < 4; ++t) {
    int n = n0 + t * 16 + l16;
    float bv = bias ? bias[n] : 0.0f;
#pragma unroll
    for (int r = 0; r < 8; ++r) {
      int m = m0 + wave * 16 + hf * 8 + r;      // C layout: VGPR r -> M = hf*8+r
      float v = acc[t][r] + bv;
      if (OUT_F32) ((float*)Cout)[(size_t)m * N + n] = v;
      else         ((bf16*)Cout)[(size_t)m * N + n] = (bf16)v;
    }
  }
}

// ---------------------------------------------------------------- routing
__global__ __launch_bounds__(256)
void routing_kernel(const float* __restrict__ hs, const float* __restrict__ Wr,
                    const float* __restrict__ br, float* __restrict__ routing) {
  __shared__ float row[HIDDEN];
  const int token = blockIdx.x;                 // b*SEQ + s
  const int b = token / SEQ, s = token % SEQ;
  for (int i = threadIdx.x; i < HIDDEN; i += 256)
    row[i] = hs[(size_t)token * HIDDEN + i];
  __syncthreads();
  const int wave = threadIdx.x >> 5, lane = threadIdx.x & 31;
  for (int h = wave; h < NH; h += 8) {
    float sum = 0.0f;
    for (int i = lane; i < HIDDEN; i += 32) sum += row[i] * Wr[(size_t)i * NH + h];
#pragma unroll
    for (int off = 16; off; off >>= 1) sum += __shfl_down(sum, off, 32);
    if (lane == 0) {
      float v = 1.0f / (1.0f + __expf(-(sum + br[h])));
      routing[((size_t)b * NH + h) * SEQ + s] = v;
    }
  }
}

// ------------------------------------------------------- top-k threshold
// Radix-select kth largest of 2048 positive floats (bit-pattern ordered).
__global__ __launch_bounds__(256)
void topk_thresh_kernel(const float* __restrict__ routing, float* __restrict__ thresh) {
  __shared__ unsigned hist[256];
  __shared__ int sel_s[2];
  const float* r = routing + (size_t)blockIdx.x * SEQ;
  unsigned prefix = 0, prefmask = 0;
  int kk = TOPK;
  for (int pass = 0; pass < 4; ++pass) {
    const int shift = 24 - 8 * pass;
    hist[threadIdx.x] = 0;
    __syncthreads();
    for (int i = threadIdx.x; i < SEQ; i += 256) {
      unsigned u = __float_as_uint(r[i]);
      if ((u & prefmask) == prefix) atomicAdd(&hist[(u >> shift) & 255], 1u);
    }
    __syncthreads();
    if (threadIdx.x == 0) {
      int acc = 0, sel = 0, rem = kk;
      for (int bin = 255; bin >= 0; --bin) {
        int c = (int)hist[bin];
        if (acc + c >= kk) { sel = bin; rem = kk - acc; break; }
        acc += c;
      }
      sel_s[0] = sel; sel_s[1] = rem;
    }
    __syncthreads();
    prefix   |= ((unsigned)sel_s[0]) << shift;
    prefmask |= 0xFFu << shift;
    kk = sel_s[1];
    __syncthreads();
  }
  if (threadIdx.x == 0) thresh[blockIdx.x] = __uint_as_float(prefix);
}

// ------------------------------------------------------- RoPE + reshape
// src (b, s, nheads*128) -> dst (b, nheads, s, 128), optional RoPE.
__global__ void rope_reshape_kernel(const bf16* __restrict__ src, bf16* __restrict__ dst,
                                    int nheads, int do_rope) {
  size_t idx = (size_t)blockIdx.x * blockDim.x + threadIdx.x;
  size_t total = (size_t)BATCH * nheads * SEQ * 64;
  if (idx >= total) return;
  int i = (int)(idx & 63);
  size_t t = idx >> 6;
  int s = (int)(t % SEQ); t /= SEQ;
  int h = (int)(t % nheads);
  int b = (int)(t / nheads);
  const bf16* sp = src + (((size_t)b * SEQ + s) * nheads + h) * HD;
  bf16* dp = dst + (((size_t)b * nheads + h) * SEQ + s) * HD;
  float x0 = (float)sp[i], x1 = (float)sp[i + 64];
  if (do_rope) {
    float inv = __powf(1.0e6f, -(float)(2 * i) / 128.0f);
    float ang = (float)s * inv;
    float sn, cs;
    __sincosf(ang, &sn, &cs);
    dp[i]      = (bf16)(x0 * cs - x1 * sn);   // q*cos + rot_half(q)*sin, j<64
    dp[i + 64] = (bf16)(x1 * cs + x0 * sn);   // j>=64
  } else {
    dp[i] = (bf16)x0; dp[i + 64] = (bf16)x1;
  }
}

// ------------------------------------------------------- flash attention
// One wave per (b, h, 16-row q tile). Online softmax with sparse mask.
__global__ __launch_bounds__(32)
void attention_kernel(const bf16* __restrict__ Qr, const bf16* __restrict__ Kr,
                      const bf16* __restrict__ Vr, const float* __restrict__ routing,
                      const float* __restrict__ thresh, bf16* __restrict__ attnO) {
  const int qt  = blockIdx.x % (SEQ / 16);
  const int h   = (blockIdx.x / (SEQ / 16)) % NH;
  const int b   = blockIdx.x / ((SEQ / 16) * NH);
  const int kvh = h >> 1;                        // repeat_interleave(n_rep=2)

  const int lane = threadIdx.x;
  const int hf = lane >> 4, l16 = lane & 15;

  const bf16*  Qb  = Qr + (((size_t)b * NH  + h)   * SEQ + qt * 16) * HD;
  const bf16*  Kb  = Kr + (((size_t)b * KVH + kvh) * SEQ) * HD;
  const bf16*  Vb  = Vr + (((size_t)b * KVH + kvh) * SEQ) * HD;
  const float* rbh = routing + ((size_t)b * NH + h) * SEQ;
  const float  th  = thresh[b * NH + h];

  // Q fragments: A layout, d split into four K=32 chunks
  v16bf qf[4];
#pragma unroll
  for (int f = 0; f < 4; ++f) {
    v8bf lo = *(const v8bf*)(Qb + (size_t)l16 * HD + f * 32 + hf * 8);
    v8bf hi = *(const v8bf*)(Qb + (size_t)l16 * HD + f * 32 + 16 + hf * 8);
    qf[f] = cat8(lo, hi);
  }

  __shared__ __align__(32) bf16 Plds[16 * 32];   // [row][tok]
  __shared__ __align__(32) bf16 Vt[128 * 32];    // [d][tok] transposed V tile

  float runmax[8], lsum[8];
  v8f o[8];
#pragma unroll
  for (int r = 0; r < 8; ++r) { runmax[r] = -3.0e38f; lsum[r] = 0.0f; }
#pragma unroll
  for (int d = 0; d < 8; ++d) o[d] = v8f{};

  const float scl = 0.08838834764831845f;        // 1/sqrt(128)

  for (int j = 0; j < SEQ; j += 32) {
    // ---- scores for two 16-token column tiles: B frags straight from K
    v8f sfr[2] = {v8f{}, v8f{}};
#pragma unroll
    for (int u = 0; u < 2; ++u) {
#pragma unroll
      for (int f = 0; f < 4; ++f) {
        const bf16* kp = Kb + (size_t)(j + u * 16 + l16) * HD + f * 32 + hf * 8;
        v16bf bfrag = cat8(*(const v8bf*)kp, *(const v8bf*)(kp + 16));
        sfr[u] = wmma_bf16(qf[f], bfrag, sfr[u]);
      }
    }

    // ---- scale + sparse mask (per kv token)
    float sv[2][8];
#pragma unroll
    for (int u = 0; u < 2; ++u) {
      const int tok = j + u * 16 + l16;
      const bool on = rbh[tok] >= th;
#pragma unroll
      for (int r = 0; r < 8; ++r)
        sv[u][r] = on ? sfr[u][r] * scl : -3.0e38f;
    }

    // ---- online softmax update (row = hf*8 + r lives across 16 lanes)
    float sc[8];
#pragma unroll
    for (int r = 0; r < 8; ++r) {
      float m2 = fmaxf(sv[0][r], sv[1][r]);
#pragma unroll
      for (int msk = 1; msk <= 8; msk <<= 1) m2 = fmaxf(m2, __shfl_xor(m2, msk, 32));
      float nm = fmaxf(runmax[r], m2);
      sc[r] = (runmax[r] > -1.0e37f) ? __expf(runmax[r] - nm) : 0.0f;
      float p0 = (sv[0][r] > -1.0e37f && nm > -1.0e37f) ? __expf(sv[0][r] - nm) : 0.0f;
      float p1 = (sv[1][r] > -1.0e37f && nm > -1.0e37f) ? __expf(sv[1][r] - nm) : 0.0f;
      float ps = p0 + p1;
#pragma unroll
      for (int msk = 1; msk <= 8; msk <<= 1) ps += __shfl_xor(ps, msk, 32);
      lsum[r] = lsum[r] * sc[r] + ps;
      runmax[r] = nm;
      Plds[(hf * 8 + r) * 32 + l16]      = (bf16)p0;
      Plds[(hf * 8 + r) * 32 + 16 + l16] = (bf16)p1;
    }
#pragma unroll
    for (int d = 0; d < 8; ++d)
#pragma unroll
      for (int r = 0; r < 8; ++r) o[d][r] *= sc[r];

    // ---- stage V tile (32 tok x 128 d) transposed into LDS
    for (int e = lane; e < 32 * 128; e += 32) {
      int tok = e >> 7, d = e & 127;
      Vt[d * 32 + tok] = Vb[(size_t)(j + tok) * HD + d];
    }
    __syncthreads();   // single-wave WG: cheap; orders LDS write->read

    // ---- O += P(16x32) @ Vtile(32x128), 8 WMMAs
    v8bf plo = *(const v8bf*)&Plds[l16 * 32 + hf * 8];
    v8bf phi = *(const v8bf*)&Plds[l16 * 32 + 16 + hf * 8];
    v16bf pfrag = cat8(plo, phi);
#pragma unroll
    for (int dt = 0; dt < 8; ++dt) {
      v16bf vfrag = cat8(*(const v8bf*)&Vt[(dt * 16 + l16) * 32 + hf * 8],
                         *(const v8bf*)&Vt[(dt * 16 + l16) * 32 + 16 + hf * 8]);
      o[dt] = wmma_bf16(pfrag, vfrag, o[dt]);
    }
    __syncthreads();
  }

  // ---- finalize: attnO (b, s, h*128 + d), bf16
#pragma unroll
  for (int r = 0; r < 8; ++r) {
    const float inv = 1.0f / fmaxf(lsum[r], 1.0e-30f);
    const size_t row = (size_t)b * SEQ + qt * 16 + hf * 8 + r;
#pragma unroll
    for (int dt = 0; dt < 8; ++dt) {
      const int col = h * HD + dt * 16 + l16;
      attnO[row * (NH * HD) + col] = (bf16)(o[dt][r] * inv);
    }
  }
}

// ---------------------------------------------------------------- launch
extern "C" void kernel_launch(void* const* d_in, const int* in_sizes, int n_in,
                              void* d_out, int out_size, void* d_ws, size_t ws_size,
                              hipStream_t stream) {
  const float* hs = (const float*)d_in[0];
  const float* Wq = (const float*)d_in[1];
  const float* bq = (const float*)d_in[2];
  const float* Wk = (const float*)d_in[3];
  const float* bk = (const float*)d_in[4];
  const float* Wv = (const float*)d_in[5];
  const float* bv = (const float*)d_in[6];
  const float* Wo = (const float*)d_in[7];
  const float* bo = (const float*)d_in[8];
  const float* Wr = (const float*)d_in[9];
  const float* br = (const float*)d_in[10];

  char* p = (char*)d_ws;
  auto alloc = [&](size_t bytes) -> void* {
    void* r = (void*)p;
    p += (bytes + 255) & ~(size_t)255;
    return r;
  };
  const size_t TOK = (size_t)BATCH * SEQ;                       // 4096
  bf16* hsb   = (bf16*)alloc(TOK * HIDDEN * 2);
  bf16* wqb   = (bf16*)alloc((size_t)HIDDEN * (NH  * HD) * 2);
  bf16* wkb   = (bf16*)alloc((size_t)HIDDEN * (KVH * HD) * 2);
  bf16* wvb   = (bf16*)alloc((size_t)HIDDEN * (KVH * HD) * 2);
  bf16* wob   = (bf16*)alloc((size_t)(NH * HD) * HIDDEN * 2);
  bf16* Qp    = (bf16*)alloc(TOK * (NH  * HD) * 2);
  bf16* Kp    = (bf16*)alloc(TOK * (KVH * HD) * 2);
  bf16* Vp    = (bf16*)alloc(TOK * (KVH * HD) * 2);
  bf16* Qrm   = (bf16*)alloc(TOK * (NH  * HD) * 2);
  bf16* Krm   = (bf16*)alloc(TOK * (KVH * HD) * 2);
  bf16* Vrm   = (bf16*)alloc(TOK * (KVH * HD) * 2);
  bf16* attnO = (bf16*)alloc(TOK * (NH * HD) * 2);
  float* routing = (float*)alloc((size_t)BATCH * NH * SEQ * 4);
  float* thresh  = (float*)alloc((size_t)BATCH * NH * 4);
  (void)ws_size; (void)n_in; (void)in_sizes; (void)out_size;

  // 1) convert to bf16
  f32_to_bf16_kernel<<<2048, 256, 0, stream>>>(hs, hsb, TOK * HIDDEN);
  f32_to_bf16_kernel<<<2048, 256, 0, stream>>>(Wq, wqb, (size_t)HIDDEN * NH * HD);
  f32_to_bf16_kernel<<<1024, 256, 0, stream>>>(Wk, wkb, (size_t)HIDDEN * KVH * HD);
  f32_to_bf16_kernel<<<1024, 256, 0, stream>>>(Wv, wvb, (size_t)HIDDEN * KVH * HD);
  f32_to_bf16_kernel<<<2048, 256, 0, stream>>>(Wo, wob, (size_t)NH * HD * HIDDEN);

  // 2) QKV projections (WMMA GEMMs, bf16 out)
  gemm_bf16_kernel<false><<<dim3((NH * HD) / 64, TOK / 128), 256, 0, stream>>>(
      hsb, wqb, bq, Qp, (int)TOK, NH * HD, HIDDEN);
  gemm_bf16_kernel<false><<<dim3((KVH * HD) / 64, TOK / 128), 256, 0, stream>>>(
      hsb, wkb, bk, Kp, (int)TOK, KVH * HD, HIDDEN);
  gemm_bf16_kernel<false><<<dim3((KVH * HD) / 64, TOK / 128), 256, 0, stream>>>(
      hsb, wvb, bv, Vp, (int)TOK, KVH * HD, HIDDEN);

  // 3) routing scores + 4) per-(b,h) top-k threshold
  routing_kernel<<<(int)TOK, 256, 0, stream>>>(hs, Wr, br, routing);
  topk_thresh_kernel<<<BATCH * NH, 256, 0, stream>>>(routing, thresh);

  // 5) RoPE + head-major reshape
  rope_reshape_kernel<<<(BATCH * NH  * SEQ * 64) / 256, 256, 0, stream>>>(Qp, Qrm, NH, 1);
  rope_reshape_kernel<<<(BATCH * KVH * SEQ * 64) / 256, 256, 0, stream>>>(Kp, Krm, KVH, 1);
  rope_reshape_kernel<<<(BATCH * KVH * SEQ * 64) / 256, 256, 0, stream>>>(Vp, Vrm, KVH, 0);

  // 6) sparse flash attention (WMMA QK^T and PV)
  attention_kernel<<<BATCH * NH * (SEQ / 16), 32, 0, stream>>>(
      Qrm, Krm, Vrm, routing, thresh, attnO);

  // 7) output projection (WMMA GEMM, f32 out + bias)
  gemm_bf16_kernel<true><<<dim3(HIDDEN / 64, TOK / 128), 256, 0, stream>>>(
      attnO, wob, bo, d_out, (int)TOK, HIDDEN, HIDDEN);
}
